// Phi3Attention_16415365005910
// MI455X (gfx1250) — matrix operand order repaired and checked by reading, not compiled, not run on hardware
//
#include <hip/hip_runtime.h>
#include <math.h>

#define S_LEN 2048
#define HID   3072
#define NH    32
#define NKV   8
#define HD    96
#define QKV_N (NH*HD + 2*NKV*HD)   // 4608
#define Q_POS (NH*HD)              // 3072 : offset of K block in qkv row
#define V_POS (Q_POS + NKV*HD)     // 3840 : offset of V block in qkv row
#define N_REP (NH/NKV)             // 4

typedef __attribute__((ext_vector_type(16))) __bf16 v16bf;
typedef __attribute__((ext_vector_type(16))) short  v16s;
typedef __attribute__((ext_vector_type(8)))  float  v8f;

union Frag   { v16s s; v16bf b; };
union FragLd { v16bf b; v16s s; int4 q[2]; };

// f32 -> bf16 (round to nearest even), raw bits
__device__ __forceinline__ short f2bf(float f) {
  union { float f; unsigned u; } c; c.f = f;
  unsigned u = c.u;
  return (short)((u + 0x7fffu + ((u >> 16) & 1u)) >> 16);
}

// Async 16B copy global -> LDS (ASYNCcnt-tracked, no VGPR data path).
// lds_off is a byte offset from the wave's LDS base (dynamic-LDS base == 0).
__device__ __forceinline__ void async_ld16(unsigned lds_off, const short* gptr) {
  asm volatile("global_load_async_to_lds_b128 %0, %1, off"
               :: "v"(lds_off), "v"(gptr) : "memory");
}
__device__ __forceinline__ void wait_async0() {
  asm volatile("s_wait_asynccnt 0" ::: "memory");
}

// A-fragment (16x32 bf16, M = lane&15). elem i -> k = 16g + i.
__device__ __forceinline__ v16bf load_a_bf(const short* base, int g) {
  FragLd f;
  f.q[0] = *reinterpret_cast<const int4*>(base + 16*g);
  f.q[1] = *reinterpret_cast<const int4*>(base + 16*g + 8);
  return f.b;
}

// B-fragment (32x16 bf16, N = lane&15). elem i -> k = i + 16g; base pre-offset.
__device__ __forceinline__ v16bf load_b_bf(const short* base) {
  FragLd f;
  f.q[0] = *reinterpret_cast<const int4*>(base);
  f.q[1] = *reinterpret_cast<const int4*>(base + 8);
  return f.b;
}

// elementwise f32 -> bf16, 4 elems/thread
__global__ void cvt_f32_bf16(const float* __restrict__ src, short* __restrict__ dst, int n4) {
  int i = blockIdx.x * blockDim.x + threadIdx.x;
  if (i >= n4) return;
  float4 f = reinterpret_cast<const float4*>(src)[i];
  short4 o;
  o.x = f2bf(f.x); o.y = f2bf(f.y); o.z = f2bf(f.z); o.w = f2bf(f.w);
  reinterpret_cast<short4*>(dst)[i] = o;
}

// C(f32) = A(bf16, MxK row-major) * W(bf16, NxK row-major)^T.
// 8 waves/block; block tile 64(M) x 128(N); wave tile 32x32 (2x2 WMMA tiles).
// Double-buffered async-LDS staging: per K-slice of 32, the block cooperatively
// copies a 64x32 A panel (4KB) + 128x32 B panel (8KB) with
// global_load_async_to_lds_b128, overlapping the next slice's copies with the
// current slice's WMMAs. Dynamic LDS: [A ping|pong 8KB][B ping|pong 16KB].
__global__ void gemm_nt_bf16(const short* __restrict__ A, const short* __restrict__ W,
                             float* __restrict__ C, int K, int N) {
  extern __shared__ short smem[];                 // 12288 shorts = 24 KB
  const int tid  = threadIdx.x;
  const int lane = tid & 31;
  const int wave = tid >> 5;
  const int g  = lane >> 4;
  const int hl = lane & 15;
  const int mB = blockIdx.y * 64;
  const int nB = blockIdx.x * 128;
  const int mw = (wave & 1) * 32;
  const int nw = (wave >> 1) * 32;

  // staging thread mapping: A = 64 rows x 4 segs x 16B ; B = 128 rows x 2 x 32B
  const int arow = tid >> 2, aseg = tid & 3;
  const int brow = tid >> 1, bseg = tid & 1;
  const short* agp = A + (size_t)(mB + arow) * K + aseg * 8;
  const short* bgp = W + (size_t)(nB + brow) * K + bseg * 16;

  // LDS offsets in shorts
  auto Aoff = [](int p) { return p * 2048; };          // 2 x 64*32
  auto Boff = [](int p) { return 4096 + p * 4096; };   // 2 x 128*32

  auto stage = [&](int t, int p) {
    unsigned la = (unsigned)(Aoff(p) + arow * 32 + aseg * 8) * 2u;
    async_ld16(la, agp + t * 32);
    unsigned lb = (unsigned)(Boff(p) + brow * 32 + bseg * 16) * 2u;
    async_ld16(lb,      bgp + t * 32);
    async_ld16(lb + 16, bgp + t * 32 + 8);
  };

  v8f acc00 = {}, acc01 = {}, acc10 = {}, acc11 = {};
  const int nk = K / 32;

  stage(0, 0);
  wait_async0();
  __syncthreads();

  for (int t = 0; t < nk; ++t) {
    const int p = t & 1;
    if (t + 1 < nk) stage(t + 1, p ^ 1);      // overlap next copies with compute

    const short* as0 = smem + Aoff(p) + (mw + hl) * 32;
    const short* as1 = smem + Aoff(p) + (mw + 16 + hl) * 32;
    const short* bs0 = smem + Boff(p) + (nw + hl) * 32;
    const short* bs1 = smem + Boff(p) + (nw + 16 + hl) * 32;
    v16bf a0 = load_a_bf(as0, g);
    v16bf a1 = load_a_bf(as1, g);
    v16bf b0 = load_b_bf(bs0 + 16 * g);
    v16bf b1 = load_b_bf(bs1 + 16 * g);
    acc00 = __builtin_amdgcn_wmma_f32_16x16x32_bf16(false, a0, false, b0, (short)0, acc00, false, false);
    acc01 = __builtin_amdgcn_wmma_f32_16x16x32_bf16(false, a0, false, b1, (short)0, acc01, false, false);
    acc10 = __builtin_amdgcn_wmma_f32_16x16x32_bf16(false, a1, false, b0, (short)0, acc10, false, false);
    acc11 = __builtin_amdgcn_wmma_f32_16x16x32_bf16(false, a1, false, b1, (short)0, acc11, false, false);

    wait_async0();                             // next slice landed
    __syncthreads();                           // all waves done reading slice t
  }

  const int m0 = mB + mw, n0 = nB + nw;
  for (int r = 0; r < 8; ++r) {
    C[(size_t)(m0      + r + 8*g) * N + n0      + hl] = acc00[r];
    C[(size_t)(m0      + r + 8*g) * N + n0 + 16 + hl] = acc01[r];
    C[(size_t)(m0 + 16 + r + 8*g) * N + n0      + hl] = acc10[r];
    C[(size_t)(m0 + 16 + r + 8*g) * N + n0 + 16 + hl] = acc11[r];
  }
}

// RoPE on Q,K of fp32 qkv; writes rotated bf16 into qkvb at same coordinates.
__global__ void rope_kernel(const float* __restrict__ qkv, short* __restrict__ qkvb) {
  const int total = S_LEN * (NH + NKV) * (HD/2);
  int idx = blockIdx.x * blockDim.x + threadIdx.x;
  if (idx >= total) return;
  int j    = idx % (HD/2);
  int t    = idx / (HD/2);
  int head = t % (NH + NKV);
  int s    = t / (NH + NKV);
  int col  = (head < NH) ? head * HD : Q_POS + (head - NH) * HD;
  float inv_freq = powf(10000.0f, -2.0f * (float)j / (float)HD);
  float fr = (float)s * inv_freq;
  float cs = cosf(fr), sn = sinf(fr);
  const float* p = qkv + (size_t)s * QKV_N + col;
  float x0 = p[j], x1 = p[j + HD/2];
  short* q = qkvb + (size_t)s * QKV_N + col;
  q[j]        = f2bf(x0 * cs - x1 * sn);
  q[j + HD/2] = f2bf(x1 * cs + x0 * sn);
}

// V transpose+convert: vtb[kvh][d][s] = bf16(qkv[s][V_POS + kvh*HD + d])
__global__ void vt_kernel(const float* __restrict__ qkv, short* __restrict__ vtb) {
  const int total = NKV * HD * S_LEN;
  int idx = blockIdx.x * blockDim.x + threadIdx.x;
  if (idx >= total) return;
  int s   = idx % S_LEN;
  int t   = idx / S_LEN;
  int d   = t % HD;
  int kvh = t / HD;
  vtb[(size_t)(kvh * HD + d) * S_LEN + s] =
      f2bf(qkv[(size_t)s * QKV_N + V_POS + kvh * HD + d]);
}

// Flash attention: 1 wave per (head, 16-query tile); 32-key tiles; causal;
// bf16 WMMA with fp32 online softmax; writes bf16 attn.
__global__ void attn_kernel(const short* __restrict__ qkvb, const short* __restrict__ vtb,
                            short* __restrict__ attnb) {
  __shared__ short Pl[16 * 32];
  const int lane = threadIdx.x;
  const int g  = lane >> 4;
  const int hl = lane & 15;
  const int h   = blockIdx.y;
  const int q0  = blockIdx.x * 16;
  const int kvh = h / N_REP;

  const short* qbase = qkvb + (size_t)(q0 + hl) * QKV_N + h * HD;
  v16bf qf[3];
  for (int c = 0; c < 3; ++c) qf[c] = load_a_bf(qbase + 32 * c, g);

  v8f acc[6] = {};
  float mrow[8], lrow[8];
  for (int r = 0; r < 8; ++r) { mrow[r] = -__builtin_inff(); lrow[r] = 0.0f; }

  const float scl = 0.10206207261596576f;   // 1/sqrt(96)
  const short* kbase0 = qkvb + Q_POS + kvh * HD;
  const int ntiles = (q0 + 16 + 31) / 32;

  for (int t = 0; t < ntiles; ++t) {
    const int kt = t * 32;
    // ---- scores: Q (16xHD) x K^T -> two 16-key C tiles ----
    v8f c0 = {}, c1 = {};
    const int key_a = kt + hl;
    const int key_b = kt + 16 + hl;
    const int ka = key_a < S_LEN - 1 ? key_a : S_LEN - 1;
    const int kb = key_b < S_LEN - 1 ? key_b : S_LEN - 1;
    const short* krowa = kbase0 + (size_t)ka * QKV_N;
    const short* krowb = kbase0 + (size_t)kb * QKV_N;
    for (int c = 0; c < 3; ++c) {
      v16bf b0 = load_b_bf(krowa + 32 * c + 16 * g);
      c0 = __builtin_amdgcn_wmma_f32_16x16x32_bf16(false, qf[c], false, b0, (short)0, c0, false, false);
      v16bf b1 = load_b_bf(krowb + 32 * c + 16 * g);
      c1 = __builtin_amdgcn_wmma_f32_16x16x32_bf16(false, qf[c], false, b1, (short)0, c1, false, false);
    }
    // ---- online softmax; rows r+8g across the 16-lane half ----
    for (int r = 0; r < 8; ++r) {
      const int q = q0 + r + 8 * g;
      float s0 = (key_a <= q) ? c0[r] * scl : -__builtin_inff();
      float s1 = (key_b <= q) ? c1[r] * scl : -__builtin_inff();
      float vmax = fmaxf(s0, s1);
      for (int off = 8; off; off >>= 1) vmax = fmaxf(vmax, __shfl_xor(vmax, off, 32));
      float mnew  = fmaxf(mrow[r], vmax);
      float scale = __expf(mrow[r] - mnew);
      float p0 = (key_a <= q) ? __expf(s0 - mnew) : 0.0f;
      float p1 = (key_b <= q) ? __expf(s1 - mnew) : 0.0f;
      float rs = p0 + p1;
      for (int off = 8; off; off >>= 1) rs += __shfl_xor(rs, off, 32);
      lrow[r] = lrow[r] * scale + rs;
      mrow[r] = mnew;
      for (int j = 0; j < 6; ++j) acc[j][r] *= scale;
      const int m = r + 8 * g;
      Pl[m * 32 + hl]      = f2bf(p0);
      Pl[m * 32 + 16 + hl] = f2bf(p1);
    }
    __syncthreads();
    // ---- P: C-layout -> A-layout via LDS ----
    Frag pf;
    for (int i = 0; i < 8; ++i) pf.s[i]     = Pl[hl * 32 + 16 * g + i];
    for (int i = 0; i < 8; ++i) pf.s[8 + i] = Pl[hl * 32 + 16 * g + 8 + i];
    __syncthreads();
    // ---- P * V from transposed V: contiguous keys per lane; run-level clamp
    // is safe (a 16-key run is either fully in-bounds or fully masked, p==0).
    int kstart = kt + 16 * g;
    if (kstart > S_LEN - 16) kstart = S_LEN - 16;
    for (int j = 0; j < 6; ++j) {
      const short* vrow = vtb + (size_t)(kvh * HD + j * 16 + hl) * S_LEN + kstart;
      v16bf vb = load_b_bf(vrow);
      acc[j] = __builtin_amdgcn_wmma_f32_16x16x32_bf16(false, pf.b, false, vb, (short)0, acc[j], false, false);
    }
  }
  // ---- normalize + write bf16 attn [s][h*HD + d] ----
  for (int r = 0; r < 8; ++r) {
    const float inv = 1.0f / lrow[r];
    const int m = q0 + r + 8 * g;
    for (int j = 0; j < 6; ++j)
      attnb[(size_t)m * HID + h * HD + j * 16 + hl] = f2bf(acc[j][r] * inv);
  }
}

extern "C" void kernel_launch(void* const* d_in, const int* in_sizes, int n_in,
                              void* d_out, int out_size, void* d_ws, size_t ws_size,
                              hipStream_t stream) {
  const float* hidden = (const float*)d_in[0];
  // d_in[1] = position_ids (arange, recomputed on device), d_in[2] = causal mask (implicit)
  const float* qkv_w  = (const float*)d_in[3];
  const float* o_w    = (const float*)d_in[4];
  float* out = (float*)d_out;

  // ---- workspace layout (bytes), with region reuse; peak ~96 MB ----
  char* ws = (char*)d_ws;
  float* qkv  = (float*)(ws);                                  // 2048*4608*4 = 37,748,736
  short* qkvb = (short*)(ws + 37748736);                       // 2048*4608*2 = 18,874,368
  short* vtb  = (short*)(ws + 56623104);                       // 8*96*2048*2 =  3,145,728
  short* hb   = (short*)(ws + 59768832);                       // 2048*3072*2 = 12,582,912
  short* wb1  = (short*)(ws + 72351744);                       // 4608*3072*2 = 28,311,552
  short* wb2  = hb;            // reuse hb(+wb1) region after GEMM1 (needs 18,874,368)
  short* attnb = (short*)ws;   // reuse fp32 qkv region after rope/vt (needs 12,582,912)

  const int lds_bytes = 24576; // 8KB A + 16KB B, double-buffered

  // 0) one-time f32 -> bf16 conversions
  cvt_f32_bf16<<<(S_LEN*HID/4 + 255)/256, 256, 0, stream>>>(hidden, hb, S_LEN*HID/4);
  cvt_f32_bf16<<<(QKV_N*HID/4 + 255)/256, 256, 0, stream>>>(qkv_w, wb1, QKV_N*HID/4);

  // 1) qkv(f32) = hidden * qkv_w^T  (2048 x 4608 x 3072), bf16 math
  gemm_nt_bf16<<<dim3(QKV_N/128, S_LEN/64), 256, lds_bytes, stream>>>(hb, wb1, qkv, HID, QKV_N);

  // 2) RoPE -> bf16 Q,K ; V -> bf16 transposed [kvh][d][s]
  {
    int tot = S_LEN * (NH + NKV) * (HD/2);
    rope_kernel<<<(tot + 255)/256, 256, 0, stream>>>(qkv, qkvb);
    int totv = NKV * HD * S_LEN;
    vt_kernel<<<(totv + 255)/256, 256, 0, stream>>>(qkv, vtb);
  }

  // 2b) o_w -> bf16 (region dead after GEMM1; stream-ordered here)
  cvt_f32_bf16<<<(HID*HID/4 + 255)/256, 256, 0, stream>>>(o_w, wb2, HID*HID/4);

  // 3) causal flash attention -> bf16 attn
  attn_kernel<<<dim3(S_LEN/16, NH), 32, 0, stream>>>(qkvb, vtb, attnb);

  // 4) out(f32) = attn * o_w^T  (2048 x 3072 x 3072), bf16 math
  gemm_nt_bf16<<<dim3(HID/128, S_LEN/64), 256, lds_bytes, stream>>>(attnb, wb2, out, HID, HID);
}
